// StackedMLP_83245056131523
// MI455X (gfx1250) — matrix operand order repaired and checked by reading, not compiled, not run on hardware
//
#include <hip/hip_runtime.h>

// ---------------------------------------------------------------------------
// Stacked per-channel MLP: [C=8][S=32768] rows, 64 ->512 ->512 ->512 ->64.
// Compute-bound (~309 GFLOP vs ~150MB HBM traffic) => bf16 WMMA path.
//   1) convert weights f32->bf16 once into d_ws (9.4 MB, L2-resident)
//   2) fused kernel: each 256-thread block (8 waves) runs 128 rows of one
//      channel through all 4 layers; activations ping-pong in 2x128KB LDS
//      (CDNA5 WGP has 320KB LDS); only final f32 rows go back to HBM.
// Inner loop: 2m x 4n register tile -> 8 v_wmma per K-step for
// 4 ds_load_b128 (A) + 8 global 16B loads (B); B reused across both m-tiles.
// ---------------------------------------------------------------------------

typedef __attribute__((ext_vector_type(16))) __bf16 v16bf;
typedef __attribute__((ext_vector_type(8)))  __bf16 v8bf;
typedef __attribute__((ext_vector_type(4)))  __bf16 v4bf;
typedef __attribute__((ext_vector_type(8)))  float  v8f;

#define ROWS   128
#define WAVES  8
#define UNITS  512
#define F_IN   64
#define F_OUT  64
#define NCH    8
#define SECT   32768

union BF16x16 { v16bf v; v8bf h[2]; };

// A fragment (16x32 bf16, ISA 7.12.2): lane L<16 holds row M=L, K 0..7 in
// vector elems 0..7 and K 16..23 in elems 8..15; lane L>=16 holds K 8..15 /
// K 24..31 of the same row.
__device__ __forceinline__ v16bf load_a_frag(const __bf16* __restrict__ base,
                                             int stride, int mbase, int kbase,
                                             int lane) {
  const __bf16* p = base + (mbase + (lane & 15)) * stride + kbase + ((lane >> 4) << 3);
  BF16x16 f;
  f.h[0] = *(const v8bf*)(p);
  f.h[1] = *(const v8bf*)(p + 16);
  return f.v;
}

// B fragment (32x16 bf16): lane's column n = lane&15; lanes 0..15 hold
// K 0..7 / K 16..23, lanes 16..31 hold K 8..15 / K 24..31 (same order as A).
// Weights are [OUT][IN] row-major, so each lane reads two 8-wide K runs
// from row n: two 16B loads.
__device__ __forceinline__ v16bf load_b_frag(const __bf16* __restrict__ W,
                                             int stride, int n, int kbase,
                                             int lane) {
  const __bf16* p = W + (size_t)n * stride + kbase + ((lane >> 4) << 3);
  BF16x16 f;
  f.h[0] = *(const v8bf*)(p);
  f.h[1] = *(const v8bf*)(p + 16);
  return f.v;
}

// Hidden layer: src (LDS, [ROWS][IN] bf16) -> dst (LDS, [ROWS][OUT] bf16).
// 2 m-tiles x 4 n-tiles per wave-group: each B fragment feeds 2 WMMAs.
template<int IN, int OUT, bool RELU>
__device__ __forceinline__ void mlp_layer_lds(const __bf16* __restrict__ src,
                                              __bf16* __restrict__ dst,
                                              const __bf16* __restrict__ W,
                                              const float* __restrict__ bias,
                                              int lane, int wave) {
  constexpr int NG = (OUT / 16) / 4;   // n-groups of 4 tiles
  constexpr int MG = ROWS / 32;        // m-groups of 2 tiles
  constexpr int GROUPS = MG * NG;      // 32 for OUT=512 (uniform over 8 waves)
  const int col0 = lane & 15;
  for (int g = wave; g < GROUPS; g += WAVES) {
    const int mbase = (g / NG) * 32;
    const int nbase = (g % NG) * 64;
    v8f acc[2][4];
#pragma unroll
    for (int t = 0; t < 4; ++t) {
      float bv = bias[nbase + t * 16 + col0];   // bias depends on N only
      acc[0][t] = (v8f){bv, bv, bv, bv, bv, bv, bv, bv};
      acc[1][t] = acc[0][t];
    }
    for (int k = 0; k < IN; k += 32) {
      v16bf a0 = load_a_frag(src, IN, mbase,      k, lane);
      v16bf a1 = load_a_frag(src, IN, mbase + 16, k, lane);
#pragma unroll
      for (int t = 0; t < 4; ++t) {
        v16bf b = load_b_frag(W, IN, nbase + t * 16 + col0, k, lane);
        acc[0][t] = __builtin_amdgcn_wmma_f32_16x16x32_bf16(
            false, a0, false, b, (short)0, acc[0][t], false, false);
        acc[1][t] = __builtin_amdgcn_wmma_f32_16x16x32_bf16(
            false, a1, false, b, (short)0, acc[1][t], false, false);
      }
    }
    // D layout: VGPR r, lane l -> row M = mbase + r + 8*(l>>4), col N = l&15
#pragma unroll
    for (int m = 0; m < 2; ++m) {
      const int rbase = mbase + m * 16 + ((lane >> 4) << 3);
#pragma unroll
      for (int t = 0; t < 4; ++t) {
        const int col = nbase + t * 16 + col0;
#pragma unroll
        for (int r = 0; r < 8; ++r) {
          float v = acc[m][t][r];
          if (RELU) v = __builtin_fmaxf(v, 0.0f);
          dst[(rbase + r) * OUT + col] = (__bf16)v;
        }
      }
    }
  }
}

// Final layer: [ROWS][512] bf16 in LDS -> f32 rows in global (no ReLU).
// OUT=64 -> 8 m-tiles x 4 n-tiles = exactly one (1m x 4n) group per wave.
__device__ __forceinline__ void mlp_layer_out(const __bf16* __restrict__ src,
                                              float* __restrict__ out,
                                              const __bf16* __restrict__ W,
                                              const float* __restrict__ bias,
                                              int lane, int wave) {
  const int mbase = wave * 16;
  const int col0  = lane & 15;
  v8f acc[4];
#pragma unroll
  for (int t = 0; t < 4; ++t) {
    float bv = bias[t * 16 + col0];
    acc[t] = (v8f){bv, bv, bv, bv, bv, bv, bv, bv};
  }
  for (int k = 0; k < UNITS; k += 32) {
    v16bf a = load_a_frag(src, UNITS, mbase, k, lane);
#pragma unroll
    for (int t = 0; t < 4; ++t) {
      v16bf b = load_b_frag(W, UNITS, t * 16 + col0, k, lane);
      acc[t] = __builtin_amdgcn_wmma_f32_16x16x32_bf16(
          false, a, false, b, (short)0, acc[t], false, false);
    }
  }
  const int rbase = mbase + ((lane >> 4) << 3);
#pragma unroll
  for (int t = 0; t < 4; ++t) {
#pragma unroll
    for (int r = 0; r < 8; ++r)
      out[(size_t)(rbase + r) * F_OUT + t * 16 + col0] = acc[t][r];
  }
}

__global__ __launch_bounds__(256) void
mlp_fused_kernel(const float* __restrict__ x,
                 const __bf16* __restrict__ w0, const __bf16* __restrict__ w1,
                 const __bf16* __restrict__ w2, const __bf16* __restrict__ w3,
                 const float* __restrict__ b0, const float* __restrict__ b1,
                 const float* __restrict__ b2, const float* __restrict__ b3,
                 float* __restrict__ out) {
  __shared__ __bf16 bufA[ROWS * UNITS];   // 128 KB
  __shared__ __bf16 bufB[ROWS * UNITS];   // 128 KB  (320KB WGP LDS on CDNA5)

  const int c       = blockIdx.y;
  const int rowbase = blockIdx.x * ROWS;
  const int tid     = threadIdx.x;
  const int lane    = tid & 31;
  const int wave    = tid >> 5;

  // Stage x tile (f32 -> bf16) into bufA as [ROWS][F_IN]
  const float4* xv = (const float4*)(x + ((size_t)c * SECT + rowbase) * F_IN);
  for (int i = tid; i < ROWS * (F_IN / 4); i += 256) {
    float4 f = xv[i];
    v4bf o = { (__bf16)f.x, (__bf16)f.y, (__bf16)f.z, (__bf16)f.w };
    *(v4bf*)(bufA + i * 4) = o;
  }
  __syncthreads();

  mlp_layer_lds<F_IN,  UNITS, true>(bufA, bufB, w0 + (size_t)c * UNITS * F_IN,
                                    b0 + c * UNITS, lane, wave);
  __syncthreads();
  mlp_layer_lds<UNITS, UNITS, true>(bufB, bufA, w1 + (size_t)c * UNITS * UNITS,
                                    b1 + c * UNITS, lane, wave);
  __syncthreads();
  mlp_layer_lds<UNITS, UNITS, true>(bufA, bufB, w2 + (size_t)c * UNITS * UNITS,
                                    b2 + c * UNITS, lane, wave);
  __syncthreads();
  mlp_layer_out(bufB, out + ((size_t)c * SECT + rowbase) * F_OUT,
                w3 + (size_t)c * F_OUT * UNITS, b3 + c * F_OUT, lane, wave);
}

// f32 -> bf16 weight conversion (vectorized x4)
__global__ __launch_bounds__(256) void
cvt_f32_to_bf16(const float4* __restrict__ src, v4bf* __restrict__ dst, int n4) {
  int i = blockIdx.x * blockDim.x + threadIdx.x;
  if (i < n4) {
    float4 f = src[i];
    dst[i] = (v4bf){ (__bf16)f.x, (__bf16)f.y, (__bf16)f.z, (__bf16)f.w };
  }
}

extern "C" void kernel_launch(void* const* d_in, const int* in_sizes, int n_in,
                              void* d_out, int out_size, void* d_ws, size_t ws_size,
                              hipStream_t stream) {
  const float* x  = (const float*)d_in[0];
  const float* w0 = (const float*)d_in[1];
  const float* w1 = (const float*)d_in[2];
  const float* w2 = (const float*)d_in[3];
  const float* w3 = (const float*)d_in[4];
  const float* b0 = (const float*)d_in[5];
  const float* b1 = (const float*)d_in[6];
  const float* b2 = (const float*)d_in[7];
  const float* b3 = (const float*)d_in[8];
  float* out = (float*)d_out;

  // bf16 weight copies in workspace (9.4 MB total, L2-resident)
  const size_t n0 = (size_t)NCH * UNITS * F_IN;    // 262144
  const size_t n1 = (size_t)NCH * UNITS * UNITS;   // 2097152
  const size_t n2 = n1;
  const size_t n3 = (size_t)NCH * F_OUT * UNITS;   // 262144
  __bf16* W0 = (__bf16*)d_ws;
  __bf16* W1 = W0 + n0;
  __bf16* W2 = W1 + n1;
  __bf16* W3 = W2 + n2;

  auto cvt = [&](const float* s, __bf16* d, size_t n) {
    int n4 = (int)(n / 4);
    cvt_f32_to_bf16<<<(n4 + 255) / 256, 256, 0, stream>>>(
        (const float4*)s, (v4bf*)d, n4);
  };
  cvt(w0, W0, n0);
  cvt(w1, W1, n1);
  cvt(w2, W2, n2);
  cvt(w3, W3, n3);

  dim3 grid(SECT / ROWS, NCH);   // 256 x 8 blocks
  mlp_fused_kernel<<<grid, 256, 0, stream>>>(x, W0, W1, W2, W3,
                                             b0, b1, b2, b3, out);
}
